// Graphormer_62612033241323
// MI455X (gfx1250) — matrix operand order repaired and checked by reading, not compiled, not run on hardware
//
#include <hip/hip_runtime.h>
#include <hip/hip_bf16.h>
#include <math.h>

typedef __attribute__((ext_vector_type(16))) _Float16 v16h;
typedef __attribute__((ext_vector_type(8)))  _Float16 v8h;
typedef __attribute__((ext_vector_type(8)))  float    v8f;

#define NNODE 2048
#define NT    2049
#define MPAD  2080   // activation row padding (multiple of 32, >= 2048+32)
#define NEDGE 32768
#define HID   512
#define HEADS 8
#define DH    64
#define FFN   2048
#define ODIM  64
#define QKVW  1536

// ---------------------------------------------------------------------------
// WMMA helpers. All matmul operands are f16-resident in memory: fragment
// loads are pure global_load_b128 of packed halves, zero cvt in the k-loop.
// ---------------------------------------------------------------------------
__device__ __forceinline__ v8f wmma16(v16h a, v16h b, v8f c) {
  return __builtin_amdgcn_wmma_f32_16x16x32_f16(false, a, false, b, (short)0, c,
                                                false, false);
}

// A-fragment: halves 0..7 <- p[0..7], halves 8..15 <- p[8..15]  (2x b128)
__device__ __forceinline__ v16h load16_gap(const _Float16* __restrict__ p) {
  v8h lo = *(const v8h*)p;
  v8h hi = *(const v8h*)(p + 8);
  return __builtin_shufflevector(lo, hi, 0, 1, 2, 3, 4, 5, 6, 7,
                                 8, 9, 10, 11, 12, 13, 14, 15);
}
// B-fragment: 16 consecutive halves along K  (2x b128)
__device__ __forceinline__ v16h load16_lin(const _Float16* __restrict__ p) {
  return *(const v16h*)p;
}

// ---------------------------------------------------------------------------
// GEMM: C[.,N] = A[.,K] @ W[N,K]^T + bias (+ReLU), f16 inputs.
// mode 0: f32 output (residual/LN consumers); mode 1: f16 output (GEMM chain).
// Grid covers padded rows exactly; N mult 64, K mult 32 -> zero guards.
// block = 256 = 8 wave32 -> 32(M) x 64(N) tile, one 16x16 WMMA per wave.
// ---------------------------------------------------------------------------
__global__ __launch_bounds__(256) void gemm16_kernel(
    const _Float16* __restrict__ A, const _Float16* __restrict__ W,
    const float* __restrict__ bias, float* __restrict__ Cf,
    _Float16* __restrict__ Ch, int N, int K, int relu, int mode) {
  int lane = threadIdx.x & 31;
  int w    = threadIdx.x >> 5;
  int n0   = blockIdx.x * 64 + (w & 3) * 16;
  int m0   = blockIdx.y * 32 + (w >> 2) * 16;
  const _Float16* pa = A + (size_t)(m0 + (lane & 15)) * K + ((lane >> 4) << 4);
  const _Float16* pb = W + (size_t)(n0 + (lane & 15)) * K + ((lane >> 4) << 4);
  v8f acc = {};
  for (int k0 = 0; k0 < K; k0 += 32) {
    __builtin_prefetch(pa + k0 + 32, 0, 1);  // global_prefetch_b8 (speculative)
    __builtin_prefetch(pb + k0 + 32, 0, 1);
    acc = wmma16(load16_gap(pa + k0), load16_lin(pb + k0), acc);
  }
  int n      = n0 + (lane & 15);
  int mb     = m0 + ((lane >> 4) << 3);
  float bval = bias[n];
#pragma unroll
  for (int v = 0; v < 8; ++v) {
    float val = acc[v] + bval;
    if (relu) val = fmaxf(val, 0.f);
    if (mode == 0) Cf[(size_t)(mb + v) * N + n] = val;
    else           Ch[(size_t)(mb + v) * N + n] = (_Float16)val;
  }
}

// ---------------------------------------------------------------------------
// Reachability in TRANSPOSED space (f16 0/1 matrices, counts exact in f32):
// Cn = S_prev @ S where S = R1^T; B[k][n] = S[k][n] = R1[n][k] -> linear loads.
// Element (m,n) of S^k is R^k[n][m] -> mark Dm[n][m] at distance t.
// ---------------------------------------------------------------------------
__global__ __launch_bounds__(256) void reach_kernel(
    const _Float16* __restrict__ Sp, const _Float16* __restrict__ R1,
    _Float16* __restrict__ Cn, unsigned char* __restrict__ Dm, int t) {
  int lane = threadIdx.x & 31;
  int w    = threadIdx.x >> 5;
  int n0   = blockIdx.x * 64 + (w & 3) * 16;
  int m0   = blockIdx.y * 32 + (w >> 2) * 16;
  const _Float16* pa = Sp + (size_t)(m0 + (lane & 15)) * NNODE + ((lane >> 4) << 4);
  const _Float16* pb = R1 + (size_t)(n0 + (lane & 15)) * NNODE + ((lane >> 4) << 4);
  v8f acc = {};
  for (int k0 = 0; k0 < NNODE; k0 += 32) {
    __builtin_prefetch(pa + k0 + 32, 0, 1);
    __builtin_prefetch(pb + k0 + 32, 0, 1);
    acc = wmma16(load16_gap(pa + k0), load16_lin(pb + k0), acc);
  }
  int n  = n0 + (lane & 15);
  int mb = m0 + ((lane >> 4) << 3);
#pragma unroll
  for (int v = 0; v < 8; ++v) {
    float r = (acc[v] > 0.5f) ? 1.f : 0.f;
    Cn[(size_t)(mb + v) * NNODE + n] = (_Float16)r;
    size_t didx = (size_t)n * NNODE + (mb + v);  // back to (i=n, j=m)
    if (r > 0.f && Dm[didx] == 5) Dm[didx] = (unsigned char)t;
  }
}

// ---------------------------------------------------------------------------
// Flash attention, one wave per (16-row tile, head). WMMA for QK^T and PV.
// qkv is f16 (MPAD rows) -> guard-free b128 loads. Bias gathered from u8 SPD.
// ---------------------------------------------------------------------------
__device__ __forceinline__ float attn_bias(const unsigned char* __restrict__ Dm,
                                           const float* __restrict__ demb,
                                           int head, int i, int j) {
  if (i >= NT || j >= NT) return -1e30f;  // mask padding (exp -> exactly 0)
  if (i == 0 || j == 0) return 0.f;       // global-token row/col
  return demb[(int)Dm[(size_t)(i - 1) * NNODE + (j - 1)] * HEADS + head];
}

__global__ __launch_bounds__(32) void attn_kernel(const _Float16* __restrict__ qkv,
                                                  const unsigned char* __restrict__ Dm,
                                                  const float* __restrict__ demb,
                                                  _Float16* __restrict__ o) {
  const int head = blockIdx.y;
  const int m0   = blockIdx.x * 16;
  const int lane = threadIdx.x;
  const int nlo  = lane & 15;
  const int hi   = lane >> 4;
  const float scale = 0.125f;  // 64^-0.5

  __shared__ _Float16 pT[16][33];

  const _Float16* pq = qkv + (size_t)(m0 + nlo) * QKVW + head * DH + (hi << 4);
  v16h qa0 = load16_gap(pq);
  v16h qa1 = load16_gap(pq + 32);

  float rM[8], rL[8];
  v8f acc[4] = {v8f{}, v8f{}, v8f{}, v8f{}};
#pragma unroll
  for (int v = 0; v < 8; ++v) { rM[v] = -1e30f; rL[v] = 0.f; }

  for (int j0 = 0; j0 < NT; j0 += 32) {
    // ---- S = Q @ K^T over two 16-col sub-tiles
    v8f sv0 = {}, sv1 = {};
    {
      const _Float16* pk0 = qkv + (size_t)(j0 + nlo) * QKVW + HID + head * DH + (hi << 4);
      const _Float16* pk1 = pk0 + (size_t)16 * QKVW;
      sv0 = wmma16(qa0, load16_lin(pk0), sv0);
      sv0 = wmma16(qa1, load16_lin(pk0 + 32), sv0);
      sv1 = wmma16(qa0, load16_lin(pk1), sv1);
      sv1 = wmma16(qa1, load16_lin(pk1 + 32), sv1);
    }
    // ---- online softmax (a row lives in 16 lanes of this half-wave)
#pragma unroll
    for (int v = 0; v < 8; ++v) {
      int i    = m0 + hi * 8 + v;
      float x0 = sv0[v] * scale + attn_bias(Dm, demb, head, i, j0 + nlo);
      float x1 = sv1[v] * scale + attn_bias(Dm, demb, head, i, j0 + 16 + nlo);
      float tm = fmaxf(x0, x1);
#pragma unroll
      for (int msk = 1; msk < 16; msk <<= 1) tm = fmaxf(tm, __shfl_xor(tm, msk, 16));
      float nm   = fmaxf(rM[v], tm);
      float corr = __expf(rM[v] - nm);
      rM[v]      = nm;
      float p0   = __expf(x0 - nm);
      float p1   = __expf(x1 - nm);
      float rs   = p0 + p1;
#pragma unroll
      for (int msk = 1; msk < 16; msk <<= 1) rs += __shfl_xor(rs, msk, 16);
      rL[v] = rL[v] * corr + rs;
#pragma unroll
      for (int t = 0; t < 4; ++t) acc[t][v] *= corr;
      pT[hi * 8 + v][nlo]      = (_Float16)p0;
      pT[hi * 8 + v][16 + nlo] = (_Float16)p1;
    }
    __syncthreads();
    // ---- P (C-layout in LDS) -> A-fragment, then O += P @ V
    v16h pa;
#pragma unroll
    for (int j = 0; j < 16; ++j) {
      int k = hi * 16 + j;
      pa[j] = pT[nlo][k];
    }
#pragma unroll
    for (int t = 0; t < 4; ++t) {
      const _Float16* pv =
          qkv + (size_t)(j0 + (hi << 4)) * QKVW + 2 * HID + head * DH + t * 16 + nlo;
      v16h bv;
#pragma unroll
      for (int j = 0; j < 16; ++j) bv[j] = pv[(size_t)j * QKVW];
      acc[t] = wmma16(pa, bv, acc[t]);
    }
    __syncthreads();
  }
  // ---- normalize and store f16 (obuf padded to MPAD rows -> no guard)
#pragma unroll
  for (int v = 0; v < 8; ++v) {
    int i     = m0 + hi * 8 + v;
    float inv = (rL[v] > 0.f) ? 1.f / rL[v] : 0.f;
#pragma unroll
    for (int t = 0; t < 4; ++t)
      o[(size_t)i * HID + head * DH + t * 16 + nlo] = (_Float16)(acc[t][v] * inv);
  }
}

// ---------------------------------------------------------------------------
// residual add + LayerNorm; writes f32 master + f16 copy for the GEMM chain
// ---------------------------------------------------------------------------
__global__ __launch_bounds__(256) void add_ln_kernel(const float* __restrict__ base,
                                                     const float* __restrict__ add,
                                                     const float* __restrict__ g,
                                                     const float* __restrict__ b,
                                                     float* __restrict__ outf,
                                                     _Float16* __restrict__ outh) {
  int row = blockIdx.x, tid = threadIdx.x;
  const float* pb = base + (size_t)row * HID;
  const float* pa = add + (size_t)row * HID;
  float x0 = pb[tid] + pa[tid];
  float x1 = pb[tid + 256] + pa[tid + 256];
  float s = x0 + x1, q = x0 * x0 + x1 * x1;
#pragma unroll
  for (int m = 1; m < 32; m <<= 1) { s += __shfl_xor(s, m, 32); q += __shfl_xor(q, m, 32); }
  __shared__ float ssum[8], ssq[8];
  int w = tid >> 5, ln = tid & 31;
  if (ln == 0) { ssum[w] = s; ssq[w] = q; }
  __syncthreads();
  if (w == 0) {
    float a2 = (ln < 8) ? ssum[ln] : 0.f;
    float q2 = (ln < 8) ? ssq[ln] : 0.f;
#pragma unroll
    for (int m = 1; m < 8; m <<= 1) { a2 += __shfl_xor(a2, m, 32); q2 += __shfl_xor(q2, m, 32); }
    if (ln == 0) { ssum[0] = a2; ssq[0] = q2; }
  }
  __syncthreads();
  float mean = ssum[0] * (1.f / HID);
  float var  = ssq[0] * (1.f / HID) - mean * mean;
  float inv  = rsqrtf(var + 1e-5f);
  float y0 = (x0 - mean) * inv * g[tid] + b[tid];
  float y1 = (x1 - mean) * inv * g[tid + 256] + b[tid + 256];
  outf[(size_t)row * HID + tid]       = y0;
  outf[(size_t)row * HID + tid + 256] = y1;
  outh[(size_t)row * HID + tid]       = (_Float16)y0;
  outh[(size_t)row * HID + tid + 256] = (_Float16)y1;
}

// ---------------------------------------------------------------------------
// log-softmax over 64 cols (one wave per row)
// ---------------------------------------------------------------------------
__global__ __launch_bounds__(32) void logsoftmax_kernel(const float* __restrict__ in,
                                                        float* __restrict__ outp) {
  int row = blockIdx.x, ln = threadIdx.x;
  float a = in[(size_t)row * ODIM + ln];
  float c = in[(size_t)row * ODIM + 32 + ln];
  float mx = fmaxf(a, c);
#pragma unroll
  for (int m = 1; m < 32; m <<= 1) mx = fmaxf(mx, __shfl_xor(mx, m, 32));
  float se = __expf(a - mx) + __expf(c - mx);
#pragma unroll
  for (int m = 1; m < 32; m <<= 1) se += __shfl_xor(se, m, 32);
  float lse = mx + __logf(se);
  outp[(size_t)row * ODIM + ln]      = a - lse;
  outp[(size_t)row * ODIM + 32 + ln] = c - lse;
}

// ---------------------------------------------------------------------------
// small utility kernels
// ---------------------------------------------------------------------------
__global__ void cvt_f16(const float* __restrict__ src, _Float16* __restrict__ dst,
                        size_t n) {
  size_t i = (size_t)blockIdx.x * blockDim.x + threadIdx.x;
  if (i < n) dst[i] = (_Float16)src[i];
}
__global__ void fill_f16(_Float16* p, float v, size_t n) {
  size_t i = (size_t)blockIdx.x * blockDim.x + threadIdx.x;
  if (i < n) p[i] = (_Float16)v;
}
__global__ void fill_f32(float* p, float v, size_t n) {
  size_t i = (size_t)blockIdx.x * blockDim.x + threadIdx.x;
  if (i < n) p[i] = v;
}
__global__ void fill_u8(unsigned char* p, unsigned char v, size_t n) {
  size_t i = (size_t)blockIdx.x * blockDim.x + threadIdx.x;
  if (i < n) p[i] = v;
}
__global__ void fill_i32(int* p, int v, int n) {
  int i = blockIdx.x * blockDim.x + threadIdx.x;
  if (i < n) p[i] = v;
}
__global__ void scatter_edges(const int* __restrict__ ei, _Float16* __restrict__ R1,
                              _Float16* __restrict__ R1T, unsigned char* __restrict__ Dm,
                              int* __restrict__ deg) {
  int e = blockIdx.x * blockDim.x + threadIdx.x;
  if (e >= NEDGE) return;
  int u = ei[e], v = ei[NEDGE + e];
  R1 [(size_t)u * NNODE + v] = (_Float16)1.f;
  R1T[(size_t)v * NNODE + u] = (_Float16)1.f;
  Dm [(size_t)u * NNODE + v] = 1;
  atomicAdd(&deg[u], 1);
}
__global__ void diag_kernel(_Float16* __restrict__ R1, _Float16* __restrict__ R1T,
                            unsigned char* __restrict__ Dm) {
  int i = blockIdx.x * blockDim.x + threadIdx.x;
  if (i < NNODE) {
    R1 [(size_t)i * NNODE + i] = (_Float16)1.f;  // R1 = I | A  (reach <= 1)
    R1T[(size_t)i * NNODE + i] = (_Float16)1.f;
    Dm [(size_t)i * NNODE + i] = 0;
  }
}
__global__ void assemble_h(const float* __restrict__ c, const int* __restrict__ deg,
                           const float* __restrict__ degemb,
                           const float* __restrict__ gtok, float* __restrict__ hf,
                           _Float16* __restrict__ hh) {
  int idx = blockIdx.x * blockDim.x + threadIdx.x;
  if (idx >= NT * HID) return;
  int row = idx / HID, col = idx % HID;
  float val;
  if (row == 0) val = gtok[col];
  else {
    int d = deg[row - 1];
    if (d > 511) d = 511;
    val = c[(size_t)(row - 1) * HID + col] + degemb[(size_t)d * HID + col];
  }
  hf[idx] = val;
  hh[idx] = (_Float16)val;
}

// ---------------------------------------------------------------------------
extern "C" void kernel_launch(void* const* d_in, const int* in_sizes, int n_in,
                              void* d_out, int out_size, void* d_ws, size_t ws_size,
                              hipStream_t stream) {
  const float* x      = (const float*)d_in[0];
  const int*   ei     = (const int*)d_in[1];
  const float* ipw    = (const float*)d_in[2];
  const float* ipb    = (const float*)d_in[3];
  const float* degemb = (const float*)d_in[4];
  const float* demb   = (const float*)d_in[5];
  const float* gtok   = (const float*)d_in[6];
  const float* qkvw   = (const float*)d_in[7];
  const float* qkvb   = (const float*)d_in[8];
  const float* aow    = (const float*)d_in[9];
  const float* aob    = (const float*)d_in[10];
  const float* f1w    = (const float*)d_in[11];
  const float* f1b    = (const float*)d_in[12];
  const float* f2w    = (const float*)d_in[13];
  const float* f2b    = (const float*)d_in[14];
  const float* l1g    = (const float*)d_in[15];
  const float* l1b    = (const float*)d_in[16];
  const float* l2g    = (const float*)d_in[17];
  const float* l2b    = (const float*)d_in[18];
  const float* opw    = (const float*)d_in[19];
  const float* opb    = (const float*)d_in[20];
  float* outp = (float*)d_out;

  // workspace layout (peak ~61 MB; everything resident in the 192 MB L2)
  char* ws          = (char*)d_ws;
  unsigned char* Dm = (unsigned char*)ws;              // [0,4M)   SPD u8 (i,j)
  int*      deg     = (int*)(ws + (4ull << 20));       // [4M,..)
  float*    hbuf    = (float*)(ws + (5ull << 20));     // [5M,9.1M)   f32 master
  _Float16* h16     = (_Float16*)(ws + (10ull << 20)); // [10M,12.1M) f16 copy
  _Float16* R1      = (_Float16*)(ws + (13ull << 20)); // [13M,21M)
  _Float16* R1T     = (_Float16*)(ws + (21ull << 20)); // [21M,29M)  S = R1^T
  _Float16* bufC    = (_Float16*)(ws + (29ull << 20)); // [29M,37M)
  _Float16* w16     = (_Float16*)(ws + (38ull << 20)); // [38M,51M) f16 weights
  float*    tmp     = (float*)(ws + (51ull << 20));    // [51M,55.1M)
  float*    big     = (float*)(ws + (56ull << 20));    // [56M,60M)
  _Float16* x16     = (_Float16*)(ws + (60ull << 20)); // [60M,60.5M)
  // post-SPD aliases
  _Float16* qkv16 = R1;                                // 6.1M <= 8M
  _Float16* o16   = R1T;                               // 2.0M <= 8M
  _Float16* f116  = bufC;                              // 8.13M (ends < 38M)
  // f16 weight sub-offsets (in halves)
  _Float16* ipw16 = w16;                    // 512*128
  _Float16* qkvw16 = ipw16 + 65536;         // 2*1536*512
  _Float16* aow16  = qkvw16 + 1572864;      // 2*512*512
  _Float16* f1w16  = aow16 + 524288;        // 2*2048*512
  _Float16* f2w16  = f1w16 + 2097152;       // 2*512*2048
  _Float16* opw16  = f2w16 + 2097152;       // 64*512
  (void)in_sizes; (void)n_in; (void)out_size; (void)ws_size;

  const size_t NN = (size_t)NNODE * NNODE;
  fill_f16<<<(NN + 255) / 256, 256, 0, stream>>>(R1, 0.f, NN);
  fill_f16<<<(NN + 255) / 256, 256, 0, stream>>>(R1T, 0.f, NN);
  fill_u8 <<<(NN + 255) / 256, 256, 0, stream>>>(Dm, (unsigned char)5, NN);
  fill_i32<<<(NNODE + 255) / 256, 256, 0, stream>>>(deg, 0, NNODE);
  // zero pad rows of h (f32 + f16) so padded GEMMs stay deterministic/finite
  size_t padn = (size_t)(MPAD - NT) * HID;
  fill_f32<<<(padn + 255) / 256, 256, 0, stream>>>(hbuf + (size_t)NT * HID, 0.f, padn);
  fill_f16<<<(padn + 255) / 256, 256, 0, stream>>>(h16 + (size_t)NT * HID, 0.f, padn);
  // one-time f16 conversion of x and all weights (~13 MB total)
  cvt_f16<<<(2048 * 128 + 255) / 256, 256, 0, stream>>>(x, x16, 2048 * 128);
  cvt_f16<<<(65536 + 255) / 256, 256, 0, stream>>>(ipw, ipw16, 65536);
  cvt_f16<<<(1572864 + 255) / 256, 256, 0, stream>>>(qkvw, qkvw16, 1572864);
  cvt_f16<<<(524288 + 255) / 256, 256, 0, stream>>>(aow, aow16, 524288);
  cvt_f16<<<(2097152 + 255) / 256, 256, 0, stream>>>(f1w, f1w16, 2097152);
  cvt_f16<<<(2097152 + 255) / 256, 256, 0, stream>>>(f2w, f2w16, 2097152);
  cvt_f16<<<(32768 + 255) / 256, 256, 0, stream>>>(opw, opw16, 32768);

  scatter_edges<<<(NEDGE + 255) / 256, 256, 0, stream>>>(ei, R1, R1T, Dm, deg);
  diag_kernel<<<(NNODE + 255) / 256, 256, 0, stream>>>(R1, R1T, Dm);

  // in_proj + degree embedding + global token
  gemm16_kernel<<<dim3(HID / 64, NNODE / 32), 256, 0, stream>>>(
      x16, ipw16, ipb, big, nullptr, HID, 128, 0, 0);
  assemble_h<<<(NT * HID + 255) / 256, 256, 0, stream>>>(big, deg, degemb, gtok,
                                                         hbuf, h16);

  // SPD via boolean matrix powers on WMMA (transposed space):
  // S^2, S^3, S^4 mark distances 2,3,4 (everything else stays 5)
  dim3 rg(NNODE / 64, NNODE / 32);
  reach_kernel<<<rg, 256, 0, stream>>>(R1T, R1, bufC, Dm, 2);  // S^2
  reach_kernel<<<rg, 256, 0, stream>>>(bufC, R1, R1T, Dm, 3);  // S^3 (R1T dead)
  reach_kernel<<<rg, 256, 0, stream>>>(R1T, R1, bufC, Dm, 4);  // S^4

  for (int l = 0; l < 2; ++l) {
    gemm16_kernel<<<dim3(QKVW / 64, MPAD / 32), 256, 0, stream>>>(
        h16, qkvw16 + (size_t)l * QKVW * HID, qkvb + (size_t)l * QKVW,
        nullptr, qkv16, QKVW, HID, 0, 1);
    attn_kernel<<<dim3(MPAD / 16, HEADS), 32, 0, stream>>>(qkv16, Dm, demb, o16);
    gemm16_kernel<<<dim3(HID / 64, MPAD / 32), 256, 0, stream>>>(
        o16, aow16 + (size_t)l * HID * HID, aob + (size_t)l * HID,
        tmp, nullptr, HID, HID, 0, 0);
    add_ln_kernel<<<NT, 256, 0, stream>>>(hbuf, tmp, l1g + l * HID, l1b + l * HID,
                                          hbuf, h16);
    gemm16_kernel<<<dim3(FFN / 64, MPAD / 32), 256, 0, stream>>>(
        h16, f1w16 + (size_t)l * FFN * HID, f1b + (size_t)l * FFN,
        nullptr, f116, FFN, HID, 1, 1);
    gemm16_kernel<<<dim3(HID / 64, MPAD / 32), 256, 0, stream>>>(
        f116, f2w16 + (size_t)l * HID * FFN, f2b + (size_t)l * HID,
        tmp, nullptr, HID, FFN, 0, 0);
    add_ln_kernel<<<NT, 256, 0, stream>>>(hbuf, tmp, l2g + l * HID, l2b + l * HID,
                                          hbuf, h16);
  }

  // out_proj on rows 1..2048 (M=2048 exact), then log-softmax
  gemm16_kernel<<<dim3(ODIM / 64, NNODE / 32), 256, 0, stream>>>(
      h16 + HID, opw16, opb, big, nullptr, ODIM, HID, 0, 0);
  logsoftmax_kernel<<<NNODE, 32, 0, stream>>>(big, outp);
}
